// Transformer_90752658964681
// MI455X (gfx1250) — matrix operand order repaired and checked by reading, not compiled, not run on hardware
//
#include <hip/hip_runtime.h>

// ---------------------------------------------------------------------------
// MI455X (gfx1250, wave32) transformer forward.
// All matmuls run through v_wmma_f32_16x16x32_bf16 (f32 accumulate).
// Main GEMM uses a double-buffered ASYNCcnt pipeline:
//   global_load_async_to_lds_b128  ->  s_wait_asynccnt 0  ->  ds_load -> WMMA
// Weights pre-transposed to bf16 [N,K] so A and B WMMA fragments are
// contiguous 16B loads per lane matching the CDNA5 wave32 VGPR layouts.
// ---------------------------------------------------------------------------

typedef __bf16 bf16;
typedef __attribute__((ext_vector_type(8)))  bf16  v8bf;
typedef __attribute__((ext_vector_type(16))) bf16  v16bf;
typedef __attribute__((ext_vector_type(8)))  float v8f;
typedef int v4i __attribute__((vector_size(16)));   // matches builtin param

#define AS1 __attribute__((address_space(1)))
#define AS3 __attribute__((address_space(3)))

#if defined(__HIP_DEVICE_COMPILE__) &&                                  \
    __has_builtin(__builtin_amdgcn_global_load_async_to_lds_b128) &&    \
    __has_builtin(__builtin_amdgcn_s_wait_asynccnt)
#define USE_ASYNC_LDS 1
// async copy of one 16B chunk: global (AS1) -> LDS (AS3)
#define GL_ASYNC_B128(gp, lp)                                           \
  __builtin_amdgcn_global_load_async_to_lds_b128(                       \
      (AS1 v4i*)(gp), (AS3 v4i*)(lp), 0, 0)
#else
#define USE_ASYNC_LDS 0
#endif

constexpr int CB  = 32;            // batch
constexpr int CS  = 512;           // sequence
constexpr int CD  = 768;           // model dim
constexpr int CH  = 12;            // heads
constexpr int CDH = 64;            // head dim
constexpr int CL  = 6;             // layers
constexpr int CC  = 16;            // classes
constexpr int CBS = CB * CS;       // 16384 tokens
constexpr int CSD = CS * CD;       // 393216
constexpr int KCH = 4096;          // classifier K-chunk
constexpr int NPART = CSD / KCH;   // 96 partials

__device__ __forceinline__ v16bf cat8(v8bf lo, v8bf hi) {
  return __builtin_shufflevector(lo, hi, 0,1,2,3,4,5,6,7,8,9,10,11,12,13,14,15);
}

__device__ __forceinline__ v8f wmma_bf16(v16bf a, v16bf b, v8f c) {
  // D = A(16x32 bf16) * B(32x16 bf16) + C(16x16 f32)
  return __builtin_amdgcn_wmma_f32_16x16x32_bf16(
      /*neg_a=*/false, a, /*neg_b=*/false, b,
      /*c_mod=*/(short)0, c, /*reuse_a=*/false, /*reuse_b=*/false);
}

// ---------------------------------------------------------------------------
// Embedding + positional encoding: x = emb*(1+sqrt(D)) + pe
// ---------------------------------------------------------------------------
__global__ void embed_kernel(const int* __restrict__ tok,
                             const float* __restrict__ emb,
                             float* __restrict__ xf, bf16* __restrict__ xb) {
  size_t idx = (size_t)blockIdx.x * blockDim.x + threadIdx.x;
  if (idx >= (size_t)CBS * CD) return;
  int d = (int)(idx % CD);
  size_t t = idx / CD;
  int s = (int)(t % CS);
  int id = tok[t];
  float e = emb[(size_t)id * CD + d];
  // ang = s / 10000^(2d/D) = s * exp(-(2d/D)*ln(10000))
  float ang = (float)s * __expf(-(2.0f * (float)d / (float)CD) * 9.210340371976184f);
  float pe = ((d & 1) == 0) ? __sinf(ang) : __cosf(ang);
  float v = e * (1.0f + 27.712812921102035f) + pe;  // 1 + sqrt(768)
  xf[idx] = v;
  xb[idx] = (bf16)v;
}

// ---------------------------------------------------------------------------
// Transpose + fp32->bf16: W[g][k][n] -> WT[g][n][k]
// ---------------------------------------------------------------------------
__global__ void transpose_bf16_kernel(const float* __restrict__ W,
                                      bf16* __restrict__ WT, int K, int N) {
  size_t g = blockIdx.y;
  size_t base = g * (size_t)K * N;
  int idx = blockIdx.x * blockDim.x + threadIdx.x;
  if (idx >= K * N) return;
  int k = idx / N, n = idx % N;
  WT[base + (size_t)n * K + k] = (bf16)W[base + idx];
}

// ---------------------------------------------------------------------------
// Generic WMMA GEMM: C[z] = A[z](M,K) * BT[z](N,K)^T + bias[z]
// 256 threads = 8 waves; block tile 32(M) x 64(N); each wave a 16x16 tile.
// Async path: K staged in 64-wide chunks through LDS (double buffered).
// Requires K % 64 == 0 (true for every call site: K = 768).
// ---------------------------------------------------------------------------
__global__ __launch_bounds__(256)
void gemm_bf16_wmma(const bf16* __restrict__ A, size_t strideA, int lda,
                    const bf16* __restrict__ BT, size_t strideBT, int ldbt,
                    const float* __restrict__ bias, size_t strideBias,
                    float* __restrict__ Cf, bf16* __restrict__ Cb,
                    size_t strideC, size_t rsC, size_t csC,
                    int M, int N, int K) {
  const int tid  = threadIdx.x;
  const int lane = tid & 31;
  const int wave = tid >> 5;
  const int wy = wave >> 2, wx = wave & 3;
  const int m0 = blockIdx.y * 32 + wy * 16;
  const int n0 = blockIdx.x * 64 + wx * 16;
  const int z  = blockIdx.z;
  const int r   = lane & 15;
  const int sel = lane >> 4;

  v8f acc = {};

#if USE_ASYNC_LDS
  // ---- async global->LDS double-buffered pipeline (ASYNCcnt) ----
  __shared__ __align__(16) bf16 Asl[2][32][64];   // 8 KB
  __shared__ __align__(16) bf16 Bsl[2][64][64];   // 16 KB
  const bf16* Ab = A  + strideA  * z;
  const bf16* Bb = BT + strideBT * z;
  // copy assignment: A tile = 256 chunks of 16B, B tile = 512 chunks of 16B
  const int ar_c = tid >> 3;             // 0..31
  const int ak_c = (tid & 7) * 8;        // 0..56 (bf16 elems)
  const int bc0  = tid >> 3;             // 0..31
  const int bc1  = (tid + 256) >> 3;     // 32..63
  const int bk_c = (tid & 7) * 8;
  bf16* ag  = const_cast<bf16*>(Ab + (size_t)(blockIdx.y * 32 + ar_c) * lda + ak_c);
  bf16* bg0 = const_cast<bf16*>(Bb + (size_t)(blockIdx.x * 64 + bc0) * ldbt + bk_c);
  bf16* bg1 = const_cast<bf16*>(Bb + (size_t)(blockIdx.x * 64 + bc1) * ldbt + bk_c);
  const int nk = K >> 6;

  // prologue: fill buffer 0
  GL_ASYNC_B128(ag,  &Asl[0][ar_c][ak_c]);
  GL_ASYNC_B128(bg0, &Bsl[0][bc0][bk_c]);
  GL_ASYNC_B128(bg1, &Bsl[0][bc1][bk_c]);

  for (int kc = 0; kc < nk; ++kc) {
    __builtin_amdgcn_s_wait_asynccnt(0);
    __syncthreads();
    const int buf = kc & 1;
    if (kc + 1 < nk) {
      const int koff = (kc + 1) << 6;
      GL_ASYNC_B128(ag  + koff, &Asl[buf ^ 1][ar_c][ak_c]);
      GL_ASYNC_B128(bg0 + koff, &Bsl[buf ^ 1][bc0][bk_c]);
      GL_ASYNC_B128(bg1 + koff, &Bsl[buf ^ 1][bc1][bk_c]);
    }
    const bf16* arL = &Asl[buf][wy * 16 + r][0];
    const bf16* bcL = &Bsl[buf][wx * 16 + r][0];
#pragma unroll
    for (int ks = 0; ks < 64; ks += 32) {
      v8bf a_lo = *(const v8bf*)(arL + ks + sel * 8);        // ds_load_b128
      v8bf a_hi = *(const v8bf*)(arL + ks + 16 + sel * 8);
      v8bf b_lo = *(const v8bf*)(bcL + ks + sel * 8);
      v8bf b_hi = *(const v8bf*)(bcL + ks + 16 + sel * 8);
      acc = wmma_bf16(cat8(a_lo, a_hi), cat8(b_lo, b_hi), acc);
    }
    __syncthreads();
  }
#else
  // ---- fallback: direct global->VGPR operand fetch ----
  const bf16* arow = A  + strideA  * z + (size_t)(m0 + r) * lda;
  const bf16* bcol = BT + strideBT * z + (size_t)(n0 + r) * ldbt;
  for (int k0 = 0; k0 < K; k0 += 32) {
    v8bf a_lo = *(const v8bf*)(arow + k0 + sel * 8);
    v8bf a_hi = *(const v8bf*)(arow + k0 + 16 + sel * 8);
    v8bf b_lo = *(const v8bf*)(bcol + k0 + sel * 8);
    v8bf b_hi = *(const v8bf*)(bcol + k0 + 16 + sel * 8);
    if (k0 + 64 < K) {
      __builtin_prefetch(arow + k0 + 64, 0, 1);   // global_prefetch_b8
      __builtin_prefetch(bcol + k0 + 64, 0, 1);
    }
    acc = wmma_bf16(cat8(a_lo, a_hi), cat8(b_lo, b_hi), acc);
  }
#endif

  const float bvv = bias ? bias[strideBias * z + n0 + r] : 0.0f;
  const int n = n0 + r;
#pragma unroll
  for (int i = 0; i < 8; ++i) {
    int m = m0 + sel * 8 + i;
    if (m < M && n < N) {
      float v = acc[i] + bvv;
      size_t idx = strideC * z + (size_t)m * rsC + (size_t)n * csC;
      if (Cf) Cf[idx] = v;
      if (Cb) Cb[idx] = (bf16)v;
    }
  }
}

// ---------------------------------------------------------------------------
// Fused attention for one (b,h, 16-row tile): scores = QK^T/8 in LDS,
// wave softmax, P*V via WMMA. q,k: [H, B*S, DH] bf16. vT: [H, DH, B*S] bf16.
// concat out: [B*S, D] bf16 (head-concat layout).
// ---------------------------------------------------------------------------
__global__ __launch_bounds__(256)
void attn_kernel(const bf16* __restrict__ qb, const bf16* __restrict__ kb,
                 const bf16* __restrict__ vtb, bf16* __restrict__ cc) {
  __shared__ __align__(16) float sc[16][CS];   // 32 KB scores
  __shared__ __align__(16) bf16  pb[16][CS];   // 16 KB probabilities
  const int lane = threadIdx.x & 31;
  const int wave = threadIdx.x >> 5;
  const int r = lane & 15, sel = lane >> 4;
  const int bh = blockIdx.y;
  const int b = bh / CH, h = bh % CH;
  const int row0 = blockIdx.x * 16;  // s base
  const size_t seg = ((size_t)h * CBS + (size_t)b * CS) * CDH;

  // Phase 1: scores tile [16, 512]; each wave covers a 64-col chunk (4 tiles)
  const bf16* arow = qb + seg + (size_t)(row0 + r) * CDH;
#pragma unroll
  for (int nt = 0; nt < 4; ++nt) {
    const int n0 = wave * 64 + nt * 16;
    const bf16* bcol = kb + seg + (size_t)(n0 + r) * CDH;
    v8f acc = {};
#pragma unroll
    for (int k0 = 0; k0 < CDH; k0 += 32) {
      v8bf a_lo = *(const v8bf*)(arow + k0 + sel * 8);
      v8bf a_hi = *(const v8bf*)(arow + k0 + 16 + sel * 8);
      v8bf b_lo = *(const v8bf*)(bcol + k0 + sel * 8);
      v8bf b_hi = *(const v8bf*)(bcol + k0 + 16 + sel * 8);
      acc = wmma_bf16(cat8(a_lo, a_hi), cat8(b_lo, b_hi), acc);
    }
#pragma unroll
    for (int i = 0; i < 8; ++i)
      sc[sel * 8 + i][n0 + r] = acc[i] * 0.125f;  // / sqrt(64)
  }
  __syncthreads();

  // Phase 2: softmax per row (8 waves x 2 rows, wave-level shuffle reduce)
#pragma unroll
  for (int rr = 0; rr < 2; ++rr) {
    const int row = wave * 2 + rr;
    float mx = -3.0e38f;
    for (int j = lane; j < CS; j += 32) mx = fmaxf(mx, sc[row][j]);
#pragma unroll
    for (int o = 16; o > 0; o >>= 1) mx = fmaxf(mx, __shfl_xor(mx, o, 32));
    float sum = 0.0f;
    for (int j = lane; j < CS; j += 32) {
      float e = __expf(sc[row][j] - mx);
      sc[row][j] = e;
      sum += e;
    }
#pragma unroll
    for (int o = 16; o > 0; o >>= 1) sum += __shfl_xor(sum, o, 32);
    const float inv = 1.0f / sum;
    for (int j = lane; j < CS; j += 32) pb[row][j] = (bf16)(sc[row][j] * inv);
  }
  __syncthreads();

  // Phase 3: out[16,64] = P[16,512] * V[512,64]; waves 0..3, one 16x16 tile each
  if (wave < 4) {
    const int n0 = wave * 16;  // dh
    const bf16* vcol = vtb + ((size_t)h * CDH + n0 + r) * CBS + (size_t)b * CS;
    v8f acc = {};
    for (int k0 = 0; k0 < CS; k0 += 32) {
      v8bf a_lo = *(const v8bf*)(&pb[r][k0 + sel * 8]);       // ds_load_b128
      v8bf a_hi = *(const v8bf*)(&pb[r][k0 + 16 + sel * 8]);
      v8bf b_lo = *(const v8bf*)(vcol + k0 + sel * 8);
      v8bf b_hi = *(const v8bf*)(vcol + k0 + 16 + sel * 8);
      acc = wmma_bf16(cat8(a_lo, a_hi), cat8(b_lo, b_hi), acc);
    }
#pragma unroll
    for (int i = 0; i < 8; ++i) {
      size_t token = (size_t)b * CS + row0 + sel * 8 + i;
      cc[token * CD + (size_t)h * CDH + n0 + r] = (bf16)acc[i];
    }
  }
}

// ---------------------------------------------------------------------------
// LayerNorm over (S,D) per batch element. t = y + x (residual).
// ---------------------------------------------------------------------------
__global__ __launch_bounds__(256)
void ln_stats_kernel(const float* __restrict__ y, const float* __restrict__ x,
                     float* __restrict__ stats) {
  __shared__ float ss[256], sq[256];
  const int b = blockIdx.x;
  const size_t base = (size_t)b * CSD;
  float s = 0.0f, q = 0.0f;
  for (int i = threadIdx.x; i < CSD; i += 256) {
    float t = y[base + i] + x[base + i];
    s += t;
    q += t * t;
  }
  ss[threadIdx.x] = s;
  sq[threadIdx.x] = q;
  __syncthreads();
  for (int o = 128; o > 0; o >>= 1) {
    if ((int)threadIdx.x < o) {
      ss[threadIdx.x] += ss[threadIdx.x + o];
      sq[threadIdx.x] += sq[threadIdx.x + o];
    }
    __syncthreads();
  }
  if (threadIdx.x == 0) {
    float mean = ss[0] / (float)CSD;
    float var  = sq[0] / (float)CSD - mean * mean;
    stats[b * 2 + 0] = mean;
    stats[b * 2 + 1] = rsqrtf(var + 1e-5f);
  }
}

__global__ void ln_apply_kernel(const float* __restrict__ y,
                                const float* __restrict__ gamma,
                                const float* __restrict__ beta,
                                const float* __restrict__ stats,
                                float* __restrict__ xf, bf16* __restrict__ xb) {
  size_t idx = (size_t)blockIdx.x * blockDim.x + threadIdx.x;
  if (idx >= (size_t)CB * CSD) return;
  size_t b = idx / CSD, rr = idx % CSD;
  float t = y[idx] + xf[idx];
  float v = (t - stats[b * 2]) * stats[b * 2 + 1] * gamma[rr] + beta[rr];
  xf[idx] = v;
  xb[idx] = (bf16)v;
}

// ---------------------------------------------------------------------------
// Classifier: logits[32,16] = x[32, 393216] * WfT[16, 393216]^T, K-split WMMA
// ---------------------------------------------------------------------------
__global__ __launch_bounds__(64)
void cls_partial_kernel(const bf16* __restrict__ xb, const bf16* __restrict__ wfT,
                        float* __restrict__ part) {
  const int lane = threadIdx.x & 31;
  const int wave = threadIdx.x >> 5;
  const int r = lane & 15, sel = lane >> 4;
  const int m0 = wave * 16;
  const int z = blockIdx.x;
  const bf16* arow = xb  + (size_t)(m0 + r) * CSD;
  const bf16* bcol = wfT + (size_t)r * CSD;
  v8f acc = {};
  const int kend = z * KCH + KCH;
  for (int k0 = z * KCH; k0 < kend; k0 += 32) {
    v8bf a_lo = *(const v8bf*)(arow + k0 + sel * 8);
    v8bf a_hi = *(const v8bf*)(arow + k0 + 16 + sel * 8);
    v8bf b_lo = *(const v8bf*)(bcol + k0 + sel * 8);
    v8bf b_hi = *(const v8bf*)(bcol + k0 + 16 + sel * 8);
    acc = wmma_bf16(cat8(a_lo, a_hi), cat8(b_lo, b_hi), acc);
  }
#pragma unroll
  for (int i = 0; i < 8; ++i) {
    int m = m0 + sel * 8 + i;
    part[(size_t)z * (CB * CC) + (size_t)m * CC + r] = acc[i];
  }
}

__global__ void cls_reduce_kernel(const float* __restrict__ part,
                                  const float* __restrict__ bias,
                                  float* __restrict__ logits) {
  int t = threadIdx.x;  // 512 = 32*16
  float s = bias[t % CC];
  for (int z = 0; z < NPART; ++z) s += part[(size_t)z * (CB * CC) + t];
  logits[t] = s;
}

__global__ void softmax_out_kernel(const float* __restrict__ logits,
                                   float* __restrict__ out) {
  int m = threadIdx.x;
  if (m >= CB) return;
  float mx = -3.0e38f;
  for (int n = 0; n < CC; ++n) mx = fmaxf(mx, logits[m * CC + n]);
  float e[CC], s = 0.0f;
  for (int n = 0; n < CC; ++n) { e[n] = __expf(logits[m * CC + n] - mx); s += e[n]; }
  float inv = 1.0f / s;
  for (int n = 0; n < CC; ++n) out[m * CC + n] = e[n] * inv;
}

// ---------------------------------------------------------------------------
extern "C" void kernel_launch(void* const* d_in, const int* in_sizes, int n_in,
                              void* d_out, int out_size, void* d_ws, size_t ws_size,
                              hipStream_t stream) {
  (void)in_sizes; (void)n_in; (void)out_size; (void)ws_size;
  const int*   tok    = (const int*)  d_in[0];
  // d_in[1] seqs_len: unused by the reference math
  const float* embedw = (const float*)d_in[2];
  const float* Wq = (const float*)d_in[3];
  const float* bq = (const float*)d_in[4];
  const float* Wk = (const float*)d_in[5];
  const float* bk = (const float*)d_in[6];
  const float* Wv = (const float*)d_in[7];
  const float* bv = (const float*)d_in[8];
  const float* Wo = (const float*)d_in[9];
  const float* bo = (const float*)d_in[10];
  const float* g1 = (const float*)d_in[11];
  const float* b1 = (const float*)d_in[12];
  const float* Wl = (const float*)d_in[13];
  const float* bl = (const float*)d_in[14];
  const float* g2 = (const float*)d_in[15];
  const float* b2 = (const float*)d_in[16];
  const float* Wf = (const float*)d_in[17];
  const float* bfv = (const float*)d_in[18];
  float* out = (float*)d_out;

  char* base = (char*)d_ws;
  size_t off = 0;
  auto alloc = [&](size_t bytes) -> void* {
    void* p = base + off;
    off += (bytes + 255) & ~(size_t)255;
    return p;
  };
  float* x_f  = (float*)alloc((size_t)CBS * CD * 4);
  bf16*  x_b  = (bf16*) alloc((size_t)CBS * CD * 2);
  float* y_f  = (float*)alloc((size_t)CBS * CD * 4);
  bf16*  q_b  = (bf16*) alloc((size_t)CH * CBS * CDH * 2);
  bf16*  k_b  = (bf16*) alloc((size_t)CH * CBS * CDH * 2);
  bf16*  vt_b = (bf16*) alloc((size_t)CH * CDH * CBS * 2);
  bf16*  cc_b = (bf16*) alloc((size_t)CBS * CD * 2);
  bf16*  wTq  = (bf16*) alloc((size_t)CH * CD * CDH * 2);
  bf16*  wTk  = (bf16*) alloc((size_t)CH * CD * CDH * 2);
  bf16*  wTv  = (bf16*) alloc((size_t)CH * CD * CDH * 2);
  bf16*  wTo  = (bf16*) alloc((size_t)CD * CD * 2);
  bf16*  wTl  = (bf16*) alloc((size_t)CD * CD * 2);
  bf16*  wfT  = (bf16*) alloc((size_t)CC * CSD * 2);
  float* stats = (float*)alloc(CB * 2 * 4);
  float* part  = (float*)alloc((size_t)NPART * CB * CC * 4);
  float* logit = (float*)alloc((size_t)CB * CC * 4);

  // Embedding + positional encoding
  {
    size_t n = (size_t)CBS * CD;
    embed_kernel<<<dim3((unsigned)((n + 255) / 256)), 256, 0, stream>>>(tok, embedw, x_f, x_b);
  }
  // Classifier weight transpose (once)
  transpose_bf16_kernel<<<dim3((CSD * CC + 255) / 256, 1), 256, 0, stream>>>(Wf, wfT, CSD, CC);

  for (int l = 0; l < CL; ++l) {
    const size_t wqkv_off = (size_t)l * CH * CD * CDH;
    const size_t wdd_off  = (size_t)l * CD * CD;
    // Weight transpose+convert for this layer
    transpose_bf16_kernel<<<dim3((CD * CDH + 255) / 256, CH), 256, 0, stream>>>(Wq + wqkv_off, wTq, CD, CDH);
    transpose_bf16_kernel<<<dim3((CD * CDH + 255) / 256, CH), 256, 0, stream>>>(Wk + wqkv_off, wTk, CD, CDH);
    transpose_bf16_kernel<<<dim3((CD * CDH + 255) / 256, CH), 256, 0, stream>>>(Wv + wqkv_off, wTv, CD, CDH);
    transpose_bf16_kernel<<<dim3((CD * CD + 255) / 256, 1), 256, 0, stream>>>(Wo + wdd_off, wTo, CD, CD);
    transpose_bf16_kernel<<<dim3((CD * CD + 255) / 256, 1), 256, 0, stream>>>(Wl + wdd_off, wTl, CD, CD);

    // Q, K: [H, B*S, DH]; V written transposed: [H, DH, B*S]
    gemm_bf16_wmma<<<dim3(CDH / 64, CBS / 32, CH), 256, 0, stream>>>(
        x_b, 0, CD, wTq, (size_t)CDH * CD, CD,
        bq + (size_t)l * CH * CDH, CDH,
        nullptr, q_b, (size_t)CBS * CDH, CDH, 1, CBS, CDH, CD);
    gemm_bf16_wmma<<<dim3(CDH / 64, CBS / 32, CH), 256, 0, stream>>>(
        x_b, 0, CD, wTk, (size_t)CDH * CD, CD,
        bk + (size_t)l * CH * CDH, CDH,
        nullptr, k_b, (size_t)CBS * CDH, CDH, 1, CBS, CDH, CD);
    gemm_bf16_wmma<<<dim3(CDH / 64, CBS / 32, CH), 256, 0, stream>>>(
        x_b, 0, CD, wTv, (size_t)CDH * CD, CD,
        bv + (size_t)l * CH * CDH, CDH,
        nullptr, vt_b, (size_t)CDH * CBS, 1, CBS, CBS, CDH, CD);

    // Fused attention -> head-concat bf16 [B*S, D]
    attn_kernel<<<dim3(CS / 16, CB * CH), 256, 0, stream>>>(q_b, k_b, vt_b, cc_b);

    // Output projection: y = concat @ Wo + bo
    gemm_bf16_wmma<<<dim3(CD / 64, CBS / 32, 1), 256, 0, stream>>>(
        cc_b, 0, CD, wTo, 0, CD, bo + (size_t)l * CD, 0,
        y_f, nullptr, 0, CD, 1, CBS, CD, CD);

    // x = LN_{S,D}(y + x) with ln1
    ln_stats_kernel<<<CB, 256, 0, stream>>>(y_f, x_f, stats);
    ln_apply_kernel<<<dim3((unsigned)(((size_t)CB * CSD + 255) / 256)), 256, 0, stream>>>(
        y_f, g1 + (size_t)l * CSD, b1 + (size_t)l * CSD, stats, x_f, x_b);

    // FFN: y = x @ Wl + bl
    gemm_bf16_wmma<<<dim3(CD / 64, CBS / 32, 1), 256, 0, stream>>>(
        x_b, 0, CD, wTl, 0, CD, bl + (size_t)l * CD, 0,
        y_f, nullptr, 0, CD, 1, CBS, CD, CD);

    // x = LN_{S,D}(y + x) with ln2
    ln_stats_kernel<<<CB, 256, 0, stream>>>(y_f, x_f, stats);
    ln_apply_kernel<<<dim3((unsigned)(((size_t)CB * CSD + 255) / 256)), 256, 0, stream>>>(
        y_f, g2 + (size_t)l * CSD, b2 + (size_t)l * CSD, stats, x_f, x_b);
  }

  // Classifier + softmax
  cls_partial_kernel<<<NPART, 64, 0, stream>>>(x_b, wfT, part);
  cls_reduce_kernel<<<1, CB * CC, 0, stream>>>(part, bfv, logit);
  softmax_out_kernel<<<1, 32, 0, stream>>>(logit, out);
}
